// GNN_64931315581287
// MI455X (gfx1250) — compile-verified, matrix operand order checked by reading, not run on hardware
//
#include <hip/hip_runtime.h>

typedef __attribute__((ext_vector_type(16))) _Float16 v16h;
typedef __attribute__((ext_vector_type(8)))  float    v8f;

#define BATCH   64
#define NP1     577
#define DMODEL  768
#define NPATCH  576
#define DENS    288
#define SNODES  288
#define GSEL    144
#define TOUT    433          // 1 + DENS + GSEL
#define AOUT    432          // DENS + GSEL
#define NWORDS  ((SNODES*SNODES)/32)   // 2592 words of bit-packed adjacency
#define MAXEDGE 1200         // >= 4*SNODES worst-case unique edges per batch

// ---------------------------------------------------------------------------
// K1: per-batch stable descending argsort of cls_attn (bitonic over 1024 in LDS)
// key = (~orderMap(val) << 32) | idx  -> ascending sort == desc val, asc idx ties
// ---------------------------------------------------------------------------
__global__ __launch_bounds__(512) void sort_kernel(const float* __restrict__ cls_attn,
                                                   int* __restrict__ order)
{
    __shared__ unsigned long long keys[1024];
    int b = blockIdx.x, tid = threadIdx.x;
    for (int i = tid; i < 1024; i += 512) {
        unsigned long long kk = ~0ull;
        if (i < NPATCH) {
            unsigned u = __float_as_uint(cls_attn[b*NPATCH + i]);
            u ^= (u >> 31) ? 0xFFFFFFFFu : 0x80000000u;   // ascending order map
            kk = ((unsigned long long)(~u) << 32) | (unsigned)i;
        }
        keys[i] = kk;
    }
    __syncthreads();
    for (int k = 2; k <= 1024; k <<= 1) {
        for (int j = k >> 1; j > 0; j >>= 1) {
            for (int i = tid; i < 1024; i += 512) {
                int l = i ^ j;
                if (l > i) {
                    bool asc = ((i & k) == 0);
                    unsigned long long a = keys[i], c = keys[l];
                    if ((a > c) == asc) { keys[i] = c; keys[l] = a; }
                }
            }
            __syncthreads();
        }
    }
    for (int i = tid; i < NPATCH; i += 512)
        order[b*NPATCH + i] = (int)(keys[i] & 0xFFFFFFFFull);
}

// ---------------------------------------------------------------------------
// K2: gather skipped expert rows + cosine-normalize (16-dim) into cn
// ---------------------------------------------------------------------------
__global__ __launch_bounds__(256) void norm_kernel(const float* __restrict__ expd,
                                                   const int* __restrict__ order,
                                                   float* __restrict__ cn)
{
    int idx = blockIdx.x * 256 + threadIdx.x;
    if (idx >= BATCH * SNODES) return;
    int b = idx / SNODES, i = idx % SNODES;
    int row = order[b*NPATCH + DENS + i] + 1;           // expert_distribution[:,1:] sorted tail
    const float* p = expd + ((size_t)b*NP1 + row) * 16;
    float v[16]; float s = 0.f;
#pragma unroll
    for (int e = 0; e < 16; ++e) { v[e] = p[e]; s += v[e]*v[e]; }
    float nrm = sqrtf(s); if (nrm < 1e-12f) nrm = 1e-12f;
    float inv = 1.0f / nrm;
    float* o = cn + (size_t)idx * 16;
#pragma unroll
    for (int e = 0; e < 16; ++e) o[e] = v[e] * inv;
}

// ---------------------------------------------------------------------------
// K3: WMMA similarity (K=16 padded to 32, f16 in / f32 acc) + per-row top-2.
// One wave per (batch, 16-row tile); row tile vs all 18 column tiles.
// ---------------------------------------------------------------------------
__global__ __launch_bounds__(32) void sim_kernel(const float* __restrict__ cn,
                                                 int* __restrict__ nbr)
{
    __shared__ float simrow[16 * SNODES];   // 18 KB
    int b = blockIdx.x, I = blockIdx.y;
    int l = threadIdx.x;
    int h = l >> 4;          // lane half
    int n = l & 15;
    const float* cnb = cn + (size_t)b * SNODES * 16;

    // A fragment: row = I*16 + (lane%16); elements 0..7 = K in [8h, 8h+8); 8..15 zero pad
    v16h afrag;
    {
        const float* pa = cnb + (I*16 + n)*16 + 8*h;
#pragma unroll
        for (int j = 0; j < 8; ++j) afrag[j] = (_Float16)pa[j];
#pragma unroll
        for (int j = 8; j < 16; ++j) afrag[j] = (_Float16)0.0f;
    }

    for (int J = 0; J < SNODES/16; ++J) {
        // B fragment: column node = J*16 + (lane%16); same K slicing (B = cn^T)
        v16h bfrag;
        const float* pb = cnb + (J*16 + n)*16 + 8*h;
#pragma unroll
        for (int j = 0; j < 8; ++j) bfrag[j] = (_Float16)pb[j];
#pragma unroll
        for (int j = 8; j < 16; ++j) bfrag[j] = (_Float16)0.0f;

        v8f c = {};
        c = __builtin_amdgcn_wmma_f32_16x16x32_f16(false, afrag, false, bfrag,
                                                   (short)0, c, false, false);
        // C layout: c[v] = sim[row = v + 8h][col = J*16 + n]
#pragma unroll
        for (int v = 0; v < 8; ++v)
            simrow[(v + 8*h)*SNODES + J*16 + n] = c[v];
    }
    __syncthreads();

    if (l < 16) {
        int grow = I*16 + l;
        float v0 = -3.0f, v1 = -3.0f;   // diagonal == -2 in reference; skip self instead
        int i0 = 0, i1 = 0;
        const float* sr = simrow + l*SNODES;
        for (int cidx = 0; cidx < SNODES; ++cidx) {
            if (cidx == grow) continue;
            float s = sr[cidx];
            if (s > v0)      { v1 = v0; i1 = i0; v0 = s; i0 = cidx; }
            else if (s > v1) { v1 = s; i1 = cidx; }
        }
        nbr[((size_t)b*SNODES + grow)*2 + 0] = i0;
        nbr[((size_t)b*SNODES + grow)*2 + 1] = i1;
    }
}

// ---------------------------------------------------------------------------
// K4: per-batch graph: undirected dedup (bit set), iterative directional
// degree filter, top-144 by (out-deg+1), in-edge CSR.
// ---------------------------------------------------------------------------
__global__ __launch_bounds__(512) void graph_kernel(const int* __restrict__ nbr,
                                                    int* __restrict__ gidx,
                                                    int* __restrict__ in_off,
                                                    int* __restrict__ in_list)
{
    __shared__ unsigned int adj[NWORDS];         // bit (u*288+v) : edge u->v
    __shared__ int deg[SNODES];
    __shared__ int cnt;
    __shared__ unsigned long long keys[512];
    __shared__ int off[SNODES + 1];
    int b = blockIdx.x, tid = threadIdx.x;

    for (int w = tid; w < NWORDS; w += 512) adj[w] = 0u;
    __syncthreads();
    if (tid < SNODES) {
        int s = tid;
        int n0 = nbr[((size_t)b*SNODES + s)*2 + 0];
        int n1 = nbr[((size_t)b*SNODES + s)*2 + 1];
        int e;
        e = n0*SNODES + s; atomicOr(&adj[e >> 5], 1u << (e & 31));
        e = s*SNODES + n0; atomicOr(&adj[e >> 5], 1u << (e & 31));
        e = n1*SNODES + s; atomicOr(&adj[e >> 5], 1u << (e & 31));
        e = s*SNODES + n1; atomicOr(&adj[e >> 5], 1u << (e & 31));
    }
    __syncthreads();

    // iterate: deg = in-degree of current set; keep edge iff deg[v] > deg[u];
    // stop when edge count is unchanged (matches reference while_loop).
    int c_last = -1;
    for (int iter = 0; iter < 1300; ++iter) {
        if (tid < SNODES) deg[tid] = 0;
        __syncthreads();
        for (int w = tid; w < NWORDS; w += 512) {
            unsigned int m = adj[w];
            while (m) {
                int bp = __ffs(m) - 1; m &= (m - 1);
                int idx = w*32 + bp;
                atomicAdd(&deg[idx % SNODES], 1);
            }
        }
        __syncthreads();
        for (int w = tid; w < NWORDS; w += 512) {
            unsigned int m = adj[w], nm = m;
            while (m) {
                int bp = __ffs(m) - 1; m &= (m - 1);
                int idx = w*32 + bp;
                int u = idx / SNODES, v = idx % SNODES;
                if (!(deg[v] > deg[u])) nm &= ~(1u << bp);
            }
            adj[w] = nm;
        }
        __syncthreads();
        if (tid == 0) cnt = 0;
        __syncthreads();
        int part = 0;
        for (int w = tid; w < NWORDS; w += 512) part += __popc(adj[w]);
        atomicAdd(&cnt, part);
        __syncthreads();
        int c = cnt;
        if (c == c_last) break;
        c_last = c;
    }
    __syncthreads();

    // out-degree -> nd = deg_src + 1
    if (tid < SNODES) deg[tid] = 0;
    __syncthreads();
    for (int w = tid; w < NWORDS; w += 512) {
        unsigned int m = adj[w];
        while (m) { int bp = __ffs(m)-1; m &= (m-1); int idx = w*32+bp;
                    atomicAdd(&deg[idx / SNODES], 1); }
    }
    __syncthreads();

    // top-144 by nd desc, ties -> lower index (lax.top_k semantics)
    keys[tid] = (tid < SNODES)
        ? ((((unsigned long long)(deg[tid] + 1)) << 16) | (unsigned)(0xFFFF - tid))
        : 0ull;
    __syncthreads();
    for (int k = 2; k <= 512; k <<= 1) {
        for (int j = k >> 1; j > 0; j >>= 1) {
            int i = tid, l = i ^ j;
            if (l > i) {
                bool dsc = ((i & k) == 0);
                unsigned long long a = keys[i], c2 = keys[l];
                if ((a < c2) == dsc) { keys[i] = c2; keys[l] = a; }   // descending sort
            }
            __syncthreads();
        }
    }
    if (tid < GSEL) gidx[b*GSEL + tid] = 0xFFFF - (int)(keys[tid] & 0xFFFFull);
    __syncthreads();

    // in-edge CSR for the aggregation kernel
    if (tid < SNODES) deg[tid] = 0;
    __syncthreads();
    for (int w = tid; w < NWORDS; w += 512) {
        unsigned int m = adj[w];
        while (m) { int bp = __ffs(m)-1; m &= (m-1); int idx = w*32+bp;
                    atomicAdd(&deg[idx % SNODES], 1); }
    }
    __syncthreads();
    if (tid == 0) {
        off[0] = 0;
        for (int v = 0; v < SNODES; ++v) off[v+1] = off[v] + deg[v];
    }
    __syncthreads();
    if (tid <= SNODES) in_off[b*(SNODES+1) + tid] = off[tid];
    if (tid < SNODES) {
        int v = tid, p = off[v];
        for (int u = 0; u < SNODES; ++u) {
            int idx = u*SNODES + v;
            if ((adj[idx >> 5] >> (idx & 31)) & 1u) in_list[b*MAXEDGE + p++] = u;
        }
    }
}

// ---------------------------------------------------------------------------
// K5a: tokens rows 0..288 (cls + kept patches), streaming gather
// ---------------------------------------------------------------------------
__global__ __launch_bounds__(256) void copy_kernel(const float* __restrict__ x,
                                                   const int* __restrict__ order,
                                                   float* __restrict__ out)
{
    int t = blockIdx.x;           // 0..288
    int b = blockIdx.y;
    int tid = threadIdx.x;
    int srcrow = (t == 0) ? 0 : (order[b*NPATCH + (t-1)] + 1);
    const float* src = x + ((size_t)b*NP1 + srcrow) * DMODEL + tid*3;
    float* dst = out + ((size_t)b*TOUT + t) * DMODEL + tid*3;
    dst[0] = src[0]; dst[1] = src[1]; dst[2] = src[2];
}

// ---------------------------------------------------------------------------
// K5b: summ rows 289..432 : avg = x[self] + sum over surviving in-neighbors
// ---------------------------------------------------------------------------
__global__ __launch_bounds__(256) void summ_kernel(const float* __restrict__ x,
                                                   const int* __restrict__ order,
                                                   const int* __restrict__ gidx,
                                                   const int* __restrict__ in_off,
                                                   const int* __restrict__ in_list,
                                                   float* __restrict__ out)
{
    int g = blockIdx.x, b = blockIdx.y, tid = threadIdx.x;
    int j = gidx[b*GSEL + g];
    const int* ob = order + b*NPATCH + DENS;
    const float* xb = x + (size_t)b*NP1*DMODEL;
    int d = tid*3;
    int row = ob[j] + 1;
    float a0 = xb[(size_t)row*DMODEL + d + 0];
    float a1 = xb[(size_t)row*DMODEL + d + 1];
    float a2 = xb[(size_t)row*DMODEL + d + 2];
    int o0 = in_off[b*(SNODES+1) + j], o1 = in_off[b*(SNODES+1) + j + 1];
    for (int e = o0; e < o1; ++e) {
        int ru = ob[in_list[b*MAXEDGE + e]] + 1;
        a0 += xb[(size_t)ru*DMODEL + d + 0];
        a1 += xb[(size_t)ru*DMODEL + d + 1];
        a2 += xb[(size_t)ru*DMODEL + d + 2];
    }
    float* dst = out + ((size_t)b*TOUT + 1 + DENS + g) * DMODEL + d;
    dst[0] = a0; dst[1] = a1; dst[2] = a2;
}

// ---------------------------------------------------------------------------
// K5c: attn output = [kept-288 sorted attn | selected-144 attn]
// ---------------------------------------------------------------------------
__global__ __launch_bounds__(448) void attn_kernel(const float* __restrict__ cls_attn,
                                                   const int* __restrict__ order,
                                                   const int* __restrict__ gidx,
                                                   float* __restrict__ out_attn)
{
    int b = blockIdx.x, i = threadIdx.x;
    if (i >= AOUT) return;
    float v;
    if (i < DENS) {
        v = cls_attn[b*NPATCH + order[b*NPATCH + i]];
    } else {
        int j = gidx[b*GSEL + (i - DENS)];
        v = cls_attn[b*NPATCH + order[b*NPATCH + DENS + j]];
    }
    out_attn[b*AOUT + i] = v;
}

// ---------------------------------------------------------------------------
extern "C" void kernel_launch(void* const* d_in, const int* in_sizes, int n_in,
                              void* d_out, int out_size, void* d_ws, size_t ws_size,
                              hipStream_t stream)
{
    const float* x        = (const float*)d_in[0];   // (64,577,768)
    const float* cls_attn = (const float*)d_in[1];   // (64,576)
    const float* expd     = (const float*)d_in[2];   // (64,577,16)
    float* out = (float*)d_out;                      // tokens then attn

    char* ws = (char*)d_ws;
    int*   order   = (int*)ws;   ws += (size_t)BATCH*NPATCH*4;          // 64*576
    float* cn      = (float*)ws; ws += (size_t)BATCH*SNODES*16*4;       // 64*288*16
    int*   nbr     = (int*)ws;   ws += (size_t)BATCH*SNODES*2*4;        // 64*288*2
    int*   gidx    = (int*)ws;   ws += (size_t)BATCH*GSEL*4;            // 64*144
    int*   in_off  = (int*)ws;   ws += (size_t)BATCH*(SNODES+1)*4;      // 64*289
    int*   in_list = (int*)ws;   ws += (size_t)BATCH*MAXEDGE*4;         // 64*1200
    float* out_attn = out + (size_t)BATCH*TOUT*DMODEL;

    sort_kernel <<<BATCH, 512, 0, stream>>>(cls_attn, order);
    norm_kernel <<<(BATCH*SNODES + 255)/256, 256, 0, stream>>>(expd, order, cn);
    sim_kernel  <<<dim3(BATCH, SNODES/16), 32, 0, stream>>>(cn, nbr);
    graph_kernel<<<BATCH, 512, 0, stream>>>(nbr, gidx, in_off, in_list);
    copy_kernel <<<dim3(1 + DENS, BATCH), 256, 0, stream>>>(x, order, out);
    summ_kernel <<<dim3(GSEL, BATCH), 256, 0, stream>>>(x, order, gidx, in_off, in_list, out);
    attn_kernel <<<BATCH, 448, 0, stream>>>(cls_attn, order, gidx, out_attn);
}
